// ContrastiveWSD_65627100283644
// MI455X (gfx1250) — matrix operand order repaired and checked by reading, not compiled, not run on hardware
//
#include <hip/hip_runtime.h>
#include <hip/hip_bf16.h>
#include <math.h>

// ---------------- problem constants (from reference) ----------------
#define Nn   50000
#define Ee   300000
#define Bb   64
#define Dd   768
#define Hh   256
#define LSs  10000
#define ETOT (Ee + Nn)       // edges + self loops
#define NEG_SLOPE 0.2f

typedef __attribute__((ext_vector_type(16))) __bf16 v16bf;
typedef __attribute__((ext_vector_type(8)))  __bf16 v8bf;
typedef __attribute__((ext_vector_type(8)))  float  v8f;

// ---------------- helpers ----------------
__device__ __forceinline__ unsigned f2ord(float f) {
    unsigned u = __float_as_uint(f);
    return (u & 0x80000000u) ? ~u : (u | 0x80000000u);
}
__device__ __forceinline__ float ord2f(unsigned u) {
    u = (u & 0x80000000u) ? (u & 0x7fffffffu) : ~u;
    return __uint_as_float(u);
}
__device__ __forceinline__ float gelu_exact(float x) {
    return 0.5f * x * (1.0f + erff(x * 0.70710678118654752f));
}

// ---------------- fused GEMM (+bias [+LN+GELU]) via WMMA bf16 ----------------
// Y[M,256] = act( X[M,K] @ W[K,256] + bias ). One block = 16 rows x 256 cols,
// 16 waves, each wave owns one 16x16 tile. K % 32 == 0, M % 16 == 0.
// Both A (X tile) and B (W k-slice, transposed) are staged in LDS as bf16 so
// each wave's fragments are aligned ds_load_b128s feeding v_wmma directly.
template <bool LN>
__global__ __launch_bounds__(512)
void gemm16_kernel(const float* __restrict__ X, const float* __restrict__ W,
                   const float* __restrict__ bias,
                   const float* __restrict__ gamma, const float* __restrict__ beta,
                   float* __restrict__ Y, int M, int K)
{
    // Row strides padded so the 16 lane base addresses map to distinct banks
    // and every fragment chunk is 16-byte aligned.
    __shared__ __align__(16) __bf16 smA[16][40];    // [m][k], 32 K used
    __shared__ __align__(16) __bf16 smB[Hh][40];    // [n][k], transposed, 32 K used
    __shared__ float smY[16][Hh + 4];               // output strip for LayerNorm

    const int tid  = threadIdx.x;
    const int wave = tid >> 5;            // 0..15 -> N tile
    const int lane = tid & 31;
    const int m0   = blockIdx.x * 16;
    const int n0   = wave * 16;
    const int lrow = lane & 15;
    const bool hi  = lane >= 16;

    const int kbaseA = hi ? 8 : 0;        // A lane-group K offset (ISA 16-bit A layout)
    const int koffB  = hi ? 8 : 0;        // B lane-group K offset
    const int nB     = n0 + lrow;         // B/C: one N column per lane

    // staging indices
    const int sr = tid >> 5;              // A staging: row 0..15
    const int sc = tid & 31;              // A staging: k 0..31

    v8f acc = {};

    for (int k0 = 0; k0 < K; k0 += 32) {
        __syncthreads();
        // ---- stage A tile: X[m0..m0+15][k0..k0+31] -> bf16 ----
        smA[sr][sc] = (__bf16)X[(size_t)(m0 + sr) * K + k0 + sc];
        // ---- stage B slice transposed: smB[c][kk] = W[k0+kk][c], packed pairs ----
        #pragma unroll
        for (int i = 0; i < 8; ++i) {
            int idx = tid + i * 512;           // 4096 k-pairs
            int cc  = idx & 255;
            int kk  = (idx >> 8) << 1;
            float w0 = W[(size_t)(k0 + kk)     * Hh + cc];
            float w1 = W[(size_t)(k0 + kk + 1) * Hh + cc];
            union { unsigned u; __bf16 h[2]; } p;
            p.h[0] = (__bf16)w0; p.h[1] = (__bf16)w1;
            *(unsigned*)&smB[cc][kk] = p.u;    // kk even -> 4B aligned
        }
        __syncthreads();

        // ---- A fragment: a[0..7]=X[m,kbase+0..7], a[8..15]=X[m,kbase+16..23] ----
        union { v16bf v; v8bf h[2]; } ua;
        ua.h[0] = *(const v8bf*)&smA[lrow][kbaseA];
        ua.h[1] = *(const v8bf*)&smA[lrow][kbaseA + 16];
        // ---- B fragment: b[0..7]=W[k0+koffB+0..7][nB], b[8..15]=W[k0+koffB+16..23][nB] ----
        union { v16bf v; v8bf h[2]; } ub;
        ub.h[0] = *(const v8bf*)&smB[nB][koffB];
        ub.h[1] = *(const v8bf*)&smB[nB][koffB + 16];

        acc = __builtin_amdgcn_wmma_f32_16x16x32_bf16(false, ua.v, false, ub.v,
                                                      (short)0, acc, false, false);
    }

    const float bv = bias[nB];

    if (!LN) {
        #pragma unroll
        for (int r = 0; r < 8; ++r) {
            int ml = r + (hi ? 8 : 0);   // C layout: VGPR r -> M = r (+8 for hi lanes)
            Y[(size_t)(m0 + ml) * Hh + nB] = acc[r] + bv;
        }
    } else {
        #pragma unroll
        for (int r = 0; r < 8; ++r) {
            int ml = r + (hi ? 8 : 0);
            smY[ml][nB] = acc[r] + bv;
        }
        __syncthreads();
        // one wave per output row: LayerNorm over 256 + exact GELU
        const int row = wave;
        float xv[8], s = 0.f, ss = 0.f;
        #pragma unroll
        for (int j = 0; j < 8; ++j) {
            float x = smY[row][lane + j * 32];
            xv[j] = x; s += x; ss += x * x;
        }
        #pragma unroll
        for (int off = 16; off > 0; off >>= 1) {
            s  += __shfl_xor(s,  off, 32);
            ss += __shfl_xor(ss, off, 32);
        }
        const float mean = s * (1.0f / Hh);
        const float var  = ss * (1.0f / Hh) - mean * mean;
        const float rstd = rsqrtf(var + 1e-5f);
        #pragma unroll
        for (int j = 0; j < 8; ++j) {
            int c = lane + j * 32;
            float x = (xv[j] - mean) * rstd * gamma[c] + beta[c];
            Y[(size_t)(m0 + row) * Hh + c] = gelu_exact(x);
        }
    }
}

// ---------------- GAT scratch init (must run every call) ----------------
__global__ __launch_bounds__(256)
void init_gat_kernel(unsigned* __restrict__ maxb, float* __restrict__ denom,
                     float* __restrict__ acc)
{
    size_t i = (size_t)blockIdx.x * blockDim.x + threadIdx.x;
    if (i < Nn) { maxb[i] = 0u; denom[i] = 0.f; }
    const size_t total  = (size_t)Nn * Hh;
    const size_t stride = (size_t)gridDim.x * blockDim.x;
    for (size_t k = i; k < total; k += stride) acc[k] = 0.f;
}

// ---------------- edge score: e = att . leakyrelu(xl[src]+xr[dst]); seg-max ----------------
__global__ __launch_bounds__(256)
void edge_score_kernel(const float* __restrict__ xl, const float* __restrict__ xr,
                       const float* __restrict__ att,
                       const int* __restrict__ src, const int* __restrict__ dst,
                       float* __restrict__ escore, unsigned* __restrict__ maxb)
{
    int widx = blockIdx.x * (blockDim.x >> 5) + (threadIdx.x >> 5);
    int lane = threadIdx.x & 31;
    if (widx >= ETOT) return;
    int s, d;
    if (widx < Ee) { s = src[widx]; d = dst[widx]; }
    else           { s = d = widx - Ee; }            // self loop
    const float* pl = xl + (size_t)s * Hh;
    const float* pr = xr + (size_t)d * Hh;
    float a = 0.f;
    #pragma unroll
    for (int j = 0; j < 8; ++j) {
        int c = lane + j * 32;
        float v = pl[c] + pr[c];
        v = v > 0.f ? v : v * NEG_SLOPE;
        a += v * att[c];
    }
    #pragma unroll
    for (int off = 16; off > 0; off >>= 1) a += __shfl_xor(a, off, 32);
    if (lane == 0) {
        escore[widx] = a;
        atomicMax(&maxb[d], f2ord(a));
    }
}

// ---------------- exp(e - max[dst]); seg-sum ----------------
__global__ __launch_bounds__(256)
void edge_exp_kernel(const float* __restrict__ escore, const unsigned* __restrict__ maxb,
                     const int* __restrict__ dst,
                     float* __restrict__ exbuf, float* __restrict__ denom)
{
    int e = blockIdx.x * blockDim.x + threadIdx.x;
    if (e >= ETOT) return;
    int d = (e < Ee) ? dst[e] : (e - Ee);
    float ex = expf(escore[e] - ord2f(maxb[d]));
    exbuf[e] = ex;
    atomicAdd(&denom[d], ex);
}

// ---------------- out[dst] += alpha * xl[src]  (4 cols per thread) ----------------
__global__ __launch_bounds__(256)
void edge_accum_kernel(const float* __restrict__ exbuf, const float* __restrict__ denom,
                       const float* __restrict__ xl,
                       const int* __restrict__ src, const int* __restrict__ dst,
                       float* __restrict__ acc)
{
    size_t t = (size_t)blockIdx.x * blockDim.x + threadIdx.x;
    size_t e = t >> 6;                 // 64 groups of 4 columns
    if (e >= ETOT) return;
    int c4 = (int)(t & 63) << 2;
    int s, d;
    if (e < Ee) { s = src[e]; d = dst[e]; }
    else        { s = d = (int)(e - Ee); }
    float alpha = exbuf[e] / denom[d];
    const float4 v = *(const float4*)(xl + (size_t)s * Hh + c4);
    float* o = acc + (size_t)d * Hh + c4;
    atomicAdd(o + 0, alpha * v.x);
    atomicAdd(o + 1, alpha * v.y);
    atomicAdd(o + 2, alpha * v.z);
    atomicAdd(o + 3, alpha * v.w);
}

// ---------------- row-wise (x + bias) -> LN -> GELU ----------------
__global__ __launch_bounds__(256)
void ln_gelu_rows_kernel(const float* __restrict__ in, const float* __restrict__ bias,
                         const float* __restrict__ gamma, const float* __restrict__ beta,
                         float* __restrict__ out, int rows)
{
    int row  = blockIdx.x * (blockDim.x >> 5) + (threadIdx.x >> 5);
    int lane = threadIdx.x & 31;
    if (row >= rows) return;
    const float* p = in + (size_t)row * Hh;
    float xv[8], s = 0.f, ss = 0.f;
    #pragma unroll
    for (int j = 0; j < 8; ++j) {
        int c = lane + j * 32;
        float x = p[c] + bias[c];
        xv[j] = x; s += x; ss += x * x;
    }
    #pragma unroll
    for (int off = 16; off > 0; off >>= 1) {
        s  += __shfl_xor(s,  off, 32);
        ss += __shfl_xor(ss, off, 32);
    }
    const float mean = s * (1.0f / Hh);
    const float var  = ss * (1.0f / Hh) - mean * mean;
    const float rstd = rsqrtf(var + 1e-5f);
    #pragma unroll
    for (int j = 0; j < 8; ++j) {
        int c = lane + j * 32;
        float x = (xv[j] - mean) * rstd * gamma[c] + beta[c];
        out[(size_t)row * Hh + c] = gelu_exact(x);
    }
}

// ---------------- host-side orchestration ----------------
extern "C" void kernel_launch(void* const* d_in, const int* in_sizes, int n_in,
                              void* d_out, int out_size, void* d_ws, size_t ws_size,
                              hipStream_t stream)
{
    const float* word_cls  = (const float*)d_in[0];
    const float* gloss_cls = (const float*)d_in[1];
    const float* Wl_word   = (const float*)d_in[2];
    const float* bl_word   = (const float*)d_in[3];
    const float* ln_w1_g   = (const float*)d_in[4];
    const float* ln_w1_b   = (const float*)d_in[5];
    const float* Wl_con    = (const float*)d_in[6];
    const float* bl_con    = (const float*)d_in[7];
    const float* ln_c1_g   = (const float*)d_in[8];
    const float* ln_c1_b   = (const float*)d_in[9];
    const float* g2_Wl     = (const float*)d_in[10];
    const float* g2_bl     = (const float*)d_in[11];
    const float* g2_Wr     = (const float*)d_in[12];
    const float* g2_br     = (const float*)d_in[13];
    const float* g2_att    = (const float*)d_in[14];
    const float* g2_bias   = (const float*)d_in[15];
    const float* ln_c2_g   = (const float*)d_in[16];
    const float* ln_c2_b   = (const float*)d_in[17];
    const float* g3_Wl     = (const float*)d_in[18];
    const float* g3_bl     = (const float*)d_in[19];
    const float* g3_Wr     = (const float*)d_in[20];
    const float* g3_br     = (const float*)d_in[21];
    const float* g3_att    = (const float*)d_in[22];
    const float* g3_bias   = (const float*)d_in[23];
    const float* ln_c3_g   = (const float*)d_in[24];
    const float* ln_c3_b   = (const float*)d_in[25];
    const int*   src       = (const int*)d_in[26];
    const int*   dst       = src + Ee;

    float* outF = (float*)d_out;

    // workspace layout
    float*    bufA  = (float*)d_ws;                    // [N,H] node features
    float*    bufXL = bufA  + (size_t)Nn * Hh;         // [N,H]
    float*    bufXR = bufXL + (size_t)Nn * Hh;         // [N,H]
    float*    bufAC = bufXR + (size_t)Nn * Hh;         // [N,H] segment-sum accumulator
    float*    eS    = bufAC + (size_t)Nn * Hh;         // [ETOT]
    float*    eX    = eS + ETOT;                       // [ETOT]
    float*    denom = eX + ETOT;                       // [N]
    unsigned* maxb  = (unsigned*)(denom + Nn);         // [N]

    const dim3 blk512(512), blk256(256);

    // word branch: [64,768]@[768,256] + bias -> LN -> GELU -> d_out[0:16384]
    gemm16_kernel<true><<<Bb / 16, blk512, 0, stream>>>(
        word_cls, Wl_word, bl_word, ln_w1_g, ln_w1_b, outF, Bb, Dd);

    // gloss branch stage 1
    gemm16_kernel<true><<<Nn / 16, blk512, 0, stream>>>(
        gloss_cls, Wl_con, bl_con, ln_c1_g, ln_c1_b, bufA, Nn, Dd);

    const int waveBlocks = (ETOT + 7) / 8;             // 8 waves/block
    const int edgeBlocks = (ETOT + 255) / 256;
    const int accBlocks  = (int)(((size_t)ETOT * 64 + 255) / 256);
    const int initBlocks = (Nn + 255) / 256;
    const int rowBlocksN = (Nn + 7) / 8;
    const int rowBlocksL = (LSs + 7) / 8;

    // ---- GAT layer 2 ----
    init_gat_kernel<<<initBlocks, blk256, 0, stream>>>(maxb, denom, bufAC);
    gemm16_kernel<false><<<Nn / 16, blk512, 0, stream>>>(
        bufA, g2_Wl, g2_bl, nullptr, nullptr, bufXL, Nn, Hh);
    gemm16_kernel<false><<<Nn / 16, blk512, 0, stream>>>(
        bufA, g2_Wr, g2_br, nullptr, nullptr, bufXR, Nn, Hh);
    edge_score_kernel<<<waveBlocks, blk256, 0, stream>>>(bufXL, bufXR, g2_att, src, dst, eS, maxb);
    edge_exp_kernel  <<<edgeBlocks, blk256, 0, stream>>>(eS, maxb, dst, eX, denom);
    edge_accum_kernel<<<accBlocks,  blk256, 0, stream>>>(eX, denom, bufXL, src, dst, bufAC);
    ln_gelu_rows_kernel<<<rowBlocksN, blk256, 0, stream>>>(bufAC, g2_bias, ln_c2_g, ln_c2_b, bufA, Nn);

    // ---- GAT layer 3 ----
    init_gat_kernel<<<initBlocks, blk256, 0, stream>>>(maxb, denom, bufAC);
    gemm16_kernel<false><<<Nn / 16, blk512, 0, stream>>>(
        bufA, g3_Wl, g3_bl, nullptr, nullptr, bufXL, Nn, Hh);
    gemm16_kernel<false><<<Nn / 16, blk512, 0, stream>>>(
        bufA, g3_Wr, g3_br, nullptr, nullptr, bufXR, Nn, Hh);
    edge_score_kernel<<<waveBlocks, blk256, 0, stream>>>(bufXL, bufXR, g3_att, src, dst, eS, maxb);
    edge_exp_kernel  <<<edgeBlocks, blk256, 0, stream>>>(eS, maxb, dst, eX, denom);
    edge_accum_kernel<<<accBlocks,  blk256, 0, stream>>>(eX, denom, bufXL, src, dst, bufAC);
    // final LN+GELU only for labeled rows, straight into d_out
    ln_gelu_rows_kernel<<<rowBlocksL, blk256, 0, stream>>>(
        bufAC, g3_bias, ln_c3_g, ln_c3_b, outF + (size_t)Bb * Hh, LSs);

    (void)in_sizes; (void)n_in; (void)out_size; (void)ws_size;
}
